// RWKV_TimeMix_88270167867444
// MI455X (gfx1250) — compile-verified, matrix operand order checked by reading, not run on hardware
//
#include <hip/hip_runtime.h>
#include <hip/hip_bf16.h>

typedef __attribute__((ext_vector_type(16))) __bf16 v16bf;
typedef __attribute__((ext_vector_type(8)))  __bf16 v8bf;
typedef __attribute__((ext_vector_type(8)))  float  v8f;

#define B_ 16
#define T_ 1023
#define C_ 2048
#define M_ (B_ * T_)

#define BM 128
#define BN 128
#define BK 32
#define LDT (BK + 8)   // pad LDS rows (row stride 80B, 16B aligned) to dodge bank conflicts

static __device__ __forceinline__ int imin(int a, int b) { return a < b ? a : b; }

// ---- CDNA5 async global->LDS staging (ASYNCcnt path), with safe fallback ----
#if __has_builtin(__builtin_amdgcn_global_load_async_to_lds_b128)
#define HAVE_ASYNC_LDS 1
typedef __attribute__((vector_size(16))) int v4i_;
typedef __attribute__((address_space(1))) v4i_ g_v4i;
typedef __attribute__((address_space(3))) v4i_ l_v4i;
static __device__ __forceinline__ void async_b128(const void* g, void* l) {
  __builtin_amdgcn_global_load_async_to_lds_b128(
      (g_v4i*)(v4i_*)(void*)g, (l_v4i*)(v4i_*)l, 0, 0);
}
static __device__ __forceinline__ void wait_async0() {
#if __has_builtin(__builtin_amdgcn_s_wait_asynccnt)
  __builtin_amdgcn_s_wait_asynccnt(0);
#else
  asm volatile("s_wait_asynccnt 0" ::: "memory");
#endif
}
#else
#define HAVE_ASYNC_LDS 0
#endif

// ---------------- f32 -> bf16 conversion (weights) ----------------
__global__ void k_cvt_bf16(const float* __restrict__ src, __bf16* __restrict__ dst, int n) {
  for (int i = blockIdx.x * blockDim.x + threadIdx.x; i < n; i += gridDim.x * blockDim.x)
    dst[i] = (__bf16)src[i];
}

// ---------------- time-mix: xm = x*tm + shift(x)*(1-tm), bf16 out ----------------
__global__ void k_timemix(const float* __restrict__ x, const float* __restrict__ xx,
                          const float* __restrict__ tmix, __bf16* __restrict__ xm) {
  const int n = M_ * C_;
  for (int i = blockIdx.x * blockDim.x + threadIdx.x; i < n; i += gridDim.x * blockDim.x) {
    int c  = i & (C_ - 1);
    int bt = i >> 11;              // / C_
    int t  = bt % T_;
    int b  = bt / T_;
    float xs = (t == 0) ? xx[b * C_ + c] : x[i - C_];
    float tm = tmix[c];
    xm[i] = (__bf16)(x[i] * tm + xs * (1.0f - tm));
  }
}

// ---------------- WMMA bf16 GEMM:  O = A[MxK] @ W^T  (W is [NxK] row-major) -------------
// Block tile 128x128, K-step 32, double-buffered LDS, 8 waves = 4(M) x 2(N),
// each wave computes 2x4 = 8 v_wmma_f32_16x16x32_bf16 tiles per K-step.
// Tile staging uses GLOBAL_LOAD_ASYNC_TO_LDS_B128 (ASYNCcnt) when available.

// Fragment load per CDNA5 16-bit A/B lane layout: lanes 0-15 row r hold K=0..7 & 16..23,
// lanes 16-31 hold K=8..15 & 24..31 -> two 16-byte LDS chunks per lane.
static __device__ __forceinline__ v16bf load_frag_lds(const __bf16* base, int lane) {
  int row = lane & 15;
  int k0  = (lane >> 4) << 3;                 // 0 or 8
  const __bf16* p = base + row * LDT + k0;
  v8bf c0 = *(const v8bf*)(p);
  v8bf c1 = *(const v8bf*)(p + 16);
  v16bf f;
#pragma unroll
  for (int i = 0; i < 8; ++i) { f[i] = c0[i]; f[i + 8] = c1[i]; }
  return f;
}

__global__ __launch_bounds__(256) void k_gemm_bf16(
    const __bf16* __restrict__ A,
    const __bf16* __restrict__ W0, const __bf16* __restrict__ W1, const __bf16* __restrict__ W2,
    float* __restrict__ O0, float* __restrict__ O1, float* __restrict__ O2,
    int M, int N, int K) {
  const __bf16* W = (blockIdx.z == 0) ? W0 : ((blockIdx.z == 1) ? W1 : W2);
  float*        O = (blockIdx.z == 0) ? O0 : ((blockIdx.z == 1) ? O1 : O2);

  __shared__ __align__(16) __bf16 sA[2][BM][LDT];
  __shared__ __align__(16) __bf16 sB[2][BN][LDT];

  const int tid  = threadIdx.x;
  const int lane = tid & 31;
  const int wave = tid >> 5;
  const int wm   = wave >> 1;        // 0..3 -> 32 rows each
  const int wn   = wave & 1;         // 0..1 -> 64 cols each
  const int bm   = blockIdx.y * BM;
  const int bn   = blockIdx.x * BN;

  v8f acc[2][4];
#pragma unroll
  for (int i = 0; i < 2; ++i)
#pragma unroll
    for (int j = 0; j < 4; ++j) acc[i][j] = (v8f)(0.0f);

  // 256 threads stage a 128x32 bf16 tile: each thread moves 16 elements (two 16B chunks)
  auto loadTiles = [&](int kk, int buf) {
    int r  = tid >> 1;
    int hc = (tid & 1) * 16;
    int ga = imin(bm + r, M - 1);
    const __bf16* ap = A + (size_t)ga * K + kk + hc;
    const __bf16* wp = W + (size_t)(bn + r) * K + kk + hc;
#if HAVE_ASYNC_LDS
    async_b128(ap,     &sA[buf][r][hc]);
    async_b128(ap + 8, &sA[buf][r][hc + 8]);
    async_b128(wp,     &sB[buf][r][hc]);
    async_b128(wp + 8, &sB[buf][r][hc + 8]);
#else
    *(v8bf*)&sA[buf][r][hc]     = *(const v8bf*)(ap);
    *(v8bf*)&sA[buf][r][hc + 8] = *(const v8bf*)(ap + 8);
    *(v8bf*)&sB[buf][r][hc]     = *(const v8bf*)(wp);
    *(v8bf*)&sB[buf][r][hc + 8] = *(const v8bf*)(wp + 8);
#endif
    if (kk + 2 * BK < K) {                     // global_prefetch_b8 two K-steps ahead
      __builtin_prefetch(ap + 2 * BK, 0, 1);
      __builtin_prefetch(wp + 2 * BK, 0, 1);
    }
  };

  loadTiles(0, 0);
#if HAVE_ASYNC_LDS
  wait_async0();
#endif
  __syncthreads();

  const int nk = K / BK;
  for (int kb = 0; kb < nk; ++kb) {
    int buf = kb & 1;
    if (kb + 1 < nk) loadTiles((kb + 1) * BK, buf ^ 1);

    v16bf af[2], bfr[4];
#pragma unroll
    for (int i = 0; i < 2; ++i)
      af[i] = load_frag_lds(&sA[buf][wm * 32 + i * 16][0], lane);
#pragma unroll
    for (int j = 0; j < 4; ++j)
      bfr[j] = load_frag_lds(&sB[buf][wn * 64 + j * 16][0], lane);

#pragma unroll
    for (int i = 0; i < 2; ++i)
#pragma unroll
      for (int j = 0; j < 4; ++j)
        acc[i][j] = __builtin_amdgcn_wmma_f32_16x16x32_bf16(
            false, af[i], false, bfr[j], (short)0, acc[i][j], false, false);

#if HAVE_ASYNC_LDS
    wait_async0();                 // this wave's async copies into buf^1 are complete
#endif
    __syncthreads();               // all waves done -> buf^1 valid, buf free to overwrite
  }

  // C/D layout: VGPR v -> (M=v, lanes 0-15) / (M=v+8, lanes 16-31), N = lane&15
  const int rowBase = bm + wm * 32 + ((lane >> 4) << 3);
  const int colBase = bn + wn * 64 + (lane & 15);
#pragma unroll
  for (int i = 0; i < 2; ++i)
#pragma unroll
    for (int j = 0; j < 4; ++j)
#pragma unroll
      for (int v = 0; v < 8; ++v) {
        int row = rowBase + i * 16 + v;
        if (row < M) O[(size_t)row * N + colBase + j * 16] = acc[i][j][v];
      }
}

// ---------------- per-(b,c) running-max renorm: mm_new = ldexp(0.5, frexp(max_t k).e) ----
__global__ void k_rowmax(const float* __restrict__ kbuf, float* __restrict__ mm_new) {
  int i = blockIdx.x * blockDim.x + threadIdx.x;      // i = b*C + c
  if (i >= B_ * C_) return;
  int b = i >> 11, c = i & (C_ - 1);
  const float* p = kbuf + (size_t)b * T_ * C_ + c;
  float m = -3.402823466e38f;
  for (int t = 0; t < T_; ++t) m = fmaxf(m, p[(size_t)t * C_]);
  int e;
  (void)frexpf(m, &e);
  mm_new[i] = ldexpf(0.5f, e);
}

// ---------------- WKV linear scan over T, one lane per (b,c) channel ----------------
__global__ void k_scan(const float* __restrict__ kbuf, const float* __restrict__ vbuf,
                       const float* __restrict__ rbuf, const float* __restrict__ mm_new,
                       const float* __restrict__ tdecay, const float* __restrict__ tfirst,
                       const float* __restrict__ aa, const float* __restrict__ bb,
                       const float* __restrict__ mm, __bf16* __restrict__ rwkv) {
  int i = blockIdx.x * blockDim.x + threadIdx.x;      // i = b*C + c
  if (i >= B_ * C_) return;
  int b = i >> 11, c = i & (C_ - 1);

  float mmn = mm_new[i];
  float rsf = __expf(mm[i] - mmn);
  float dd  = expf(-expf(tdecay[c]));
  float ef  = expf(tfirst[c]);
  float Sk  = bb[i] * rsf;
  float Skv = aa[i] * rsf;

  size_t idx = (size_t)b * T_ * C_ + c;
  for (int t = 0; t < T_; ++t, idx += C_) {
    float ke  = expf(kbuf[idx] - mmn);
    float vv  = vbuf[idx];
    float kv  = ke * vv;
    float wk  = ef * ke + Sk;
    float wkv = ef * kv + Skv;
    Sk  = dd * Sk  + ke;
    Skv = dd * Skv + kv;
    float q = wkv / wk;
    q = (q != q) ? 0.0f : fminf(fmaxf(q, -3.402823466e38f), 3.402823466e38f);  // nan_to_num
    float sig = 1.0f / (1.0f + expf(-rbuf[idx]));
    rwkv[idx] = (__bf16)(sig * q);
  }
}

// ---------------- launch ----------------
extern "C" void kernel_launch(void* const* d_in, const int* in_sizes, int n_in,
                              void* d_out, int out_size, void* d_ws, size_t ws_size,
                              hipStream_t stream) {
  const float* x      = (const float*)d_in[0];
  const float* tdecay = (const float*)d_in[1];
  const float* tfirst = (const float*)d_in[2];
  const float* tmix   = (const float*)d_in[3];
  const float* Wk     = (const float*)d_in[4];
  const float* Wv     = (const float*)d_in[5];
  const float* Wr     = (const float*)d_in[6];
  const float* Wo     = (const float*)d_in[7];
  const float* xx     = (const float*)d_in[8];
  const float* aa     = (const float*)d_in[9];
  const float* bb     = (const float*)d_in[10];
  const float* mm     = (const float*)d_in[11];
  float* out = (float*)d_out;

  char* ws = (char*)d_ws;
  const size_t szMC_bf16 = (size_t)M_ * C_ * sizeof(__bf16);   // 67 MB
  const size_t szMC_f32  = (size_t)M_ * C_ * sizeof(float);    // 134 MB
  const size_t szW_bf16  = (size_t)C_ * C_ * sizeof(__bf16);   // 8.4 MB

  __bf16* xm_bf   = (__bf16*)(ws);
  float*  kbuf    = (float*)(ws + szMC_bf16);
  float*  vbuf    = (float*)(ws + szMC_bf16 + szMC_f32);
  float*  rbuf    = (float*)(ws + szMC_bf16 + 2 * szMC_f32);
  float*  mm_new  = (float*)(ws + szMC_bf16 + 3 * szMC_f32);
  __bf16* rwkv_bf = (__bf16*)(ws + szMC_bf16 + 3 * szMC_f32 + ((size_t)B_ * C_ * 4));
  __bf16* Wk_bf   = (__bf16*)((char*)rwkv_bf + szMC_bf16);
  __bf16* Wv_bf   = (__bf16*)((char*)Wk_bf + szW_bf16);
  __bf16* Wr_bf   = (__bf16*)((char*)Wv_bf + szW_bf16);
  __bf16* Wo_bf   = (__bf16*)((char*)Wr_bf + szW_bf16);

  const int nW = C_ * C_;
  k_cvt_bf16<<<2048, 256, 0, stream>>>(Wk, Wk_bf, nW);
  k_cvt_bf16<<<2048, 256, 0, stream>>>(Wv, Wv_bf, nW);
  k_cvt_bf16<<<2048, 256, 0, stream>>>(Wr, Wr_bf, nW);
  k_cvt_bf16<<<2048, 256, 0, stream>>>(Wo, Wo_bf, nW);

  k_timemix<<<8192, 256, 0, stream>>>(x, xx, tmix, xm_bf);

  dim3 gKVR(C_ / BN, (M_ + BM - 1) / BM, 3);
  k_gemm_bf16<<<gKVR, 256, 0, stream>>>(xm_bf, Wk_bf, Wv_bf, Wr_bf,
                                        kbuf, vbuf, rbuf, M_, C_, C_);

  k_rowmax<<<(B_ * C_ + 255) / 256, 256, 0, stream>>>(kbuf, mm_new);

  k_scan<<<(B_ * C_ + 255) / 256, 256, 0, stream>>>(kbuf, vbuf, rbuf, mm_new,
                                                    tdecay, tfirst, aa, bb, mm, rwkv_bf);

  dim3 gO(C_ / BN, (M_ + BM - 1) / BM, 1);
  k_gemm_bf16<<<gO, 256, 0, stream>>>(rwkv_bf, Wo_bf, Wo_bf, Wo_bf,
                                      out, out, out, M_, C_, C_);
}
